// BeAttentionGPT_25881472926377
// MI455X (gfx1250) — matrix operand order repaired and checked by reading, not compiled, not run on hardware
//
#include <hip/hip_runtime.h>
#include <hip/hip_bf16.h>

// Problem constants (from reference setup_inputs): B=8, S=2048, H=1024
#define BATCH 8
#define SEQ   2048
#define HDIM  1024

typedef __bf16 bf16_t;
typedef bf16_t v16bf __attribute__((ext_vector_type(16)));
typedef bf16_t v8bf  __attribute__((ext_vector_type(8)));
typedef bf16_t v4bf  __attribute__((ext_vector_type(4)));
typedef float  v8f   __attribute__((ext_vector_type(8)));
typedef __attribute__((address_space(3))) bf16_t lds_bf16_t;

// ---------- helpers ----------
__device__ __forceinline__ bf16_t f2bf(float f) {
  union { float f; unsigned u; } v; v.f = f;
  unsigned r = v.u + 0x7FFFu + ((v.u >> 16) & 1u);   // round-to-nearest-even
  unsigned short h = (unsigned short)(r >> 16);
  return __builtin_bit_cast(bf16_t, h);
}

__device__ __forceinline__ v8f zero8() {
  v8f z;
#pragma unroll
  for (int i = 0; i < 8; ++i) z[i] = 0.0f;
  return z;
}

// A-operand: 16x32 bf16 tile, rows row0..row0+15, K = k0..k0+31 (row-major src, ld halfs)
// ISA 7.12.2: lanes 0-15 -> M=lane, K 0-7 (v0-3) + 16-23 (v4-7); lanes 16-31 -> K 8-15 + 24-31
__device__ __forceinline__ v16bf load_frag_a(const bf16_t* __restrict__ base, int ld,
                                             int row0, int k0, int lane) {
  const bf16_t* p = base + (size_t)(row0 + (lane & 15)) * ld + k0 + ((lane >> 4) << 3);
  v8bf lo = *(const v8bf*)(p);
  v8bf hi = *(const v8bf*)(p + 16);
  v16bf r;
#pragma unroll
  for (int i = 0; i < 8; ++i) { r[i] = lo[i]; r[i + 8] = hi[i]; }
  return r;
}

// B-operand for D = A * srcT: logical B[k][n] = src[n0+n][k0+k] (src row-major, ld halfs).
// Lane L holds column n = L&15, K 0-7 + 16-23 (lanes 0-15) or K 8-15 + 24-31 (lanes 16-31).
__device__ __forceinline__ v16bf load_frag_bT(const bf16_t* __restrict__ base, int ld,
                                              int n0, int k0, int lane) {
  const bf16_t* p = base + (size_t)(n0 + (lane & 15)) * ld + k0 + ((lane >> 4) << 3);
  v8bf lo = *(const v8bf*)(p);
  v8bf hi = *(const v8bf*)(p + 16);
  v16bf r;
#pragma unroll
  for (int i = 0; i < 8; ++i) { r[i] = lo[i]; r[i + 8] = hi[i]; }
  return r;
}

__device__ __forceinline__ v8f wmma_bf16(v16bf a, v16bf b, v8f c) {
  return __builtin_amdgcn_wmma_f32_16x16x32_bf16(false, a, false, b, (short)0, c,
                                                 false, false);
}

// Async global -> LDS stage of a contiguous byte range (ISA §10.7 / §15.18:
// GLOBAL_LOAD_ASYNC_TO_LDS_B128, tracked with ASYNCcnt). Caller must barrier
// afterwards; we drain this wave's ASYNCcnt here.
__device__ __forceinline__ void stage_async(bf16_t* lds_dst, const bf16_t* gsrc,
                                            int bytes, int tid, int nthreads) {
  unsigned lds_base = (unsigned)(unsigned long long)(lds_bf16_t*)lds_dst;
  const char* g = (const char*)gsrc;
  for (int off = tid * 16; off < bytes; off += nthreads * 16) {
    unsigned laddr = lds_base + (unsigned)off;
    const char* gaddr = g + off;
    asm volatile("global_load_async_to_lds_b128 %0, %1, off"
                 :: "v"(laddr), "v"(gaddr) : "memory");
  }
  asm volatile("s_wait_asynccnt 0x0" ::: "memory");
}

// ---------- kernel 0: fp32 -> bf16 conversion (vectorized x4) ----------
__global__ __launch_bounds__(256) void cvt_f32_bf16(const float* __restrict__ src,
                                                    bf16_t* __restrict__ dst, int n4) {
  int i = blockIdx.x * blockDim.x + threadIdx.x;
  int stride = gridDim.x * blockDim.x;
  const float4* s4 = (const float4*)src;
  v4bf* d4 = (v4bf*)dst;
  for (; i < n4; i += stride) {
    float4 v = s4[i];
    v4bf o;
    o[0] = f2bf(v.x); o[1] = f2bf(v.y); o[2] = f2bf(v.z); o[3] = f2bf(v.w);
    d4[i] = o;
  }
}

// ---------- kernel 1: QKV projection GEMM (2x-unrolled pipeline) ----------
// C[m,n] = sum_h xb[m,h] * W[n,h] + bias[n]   (Linear: x @ W.T + b)
// z = 0 -> Qb [M,H] bf16 ; z = 1 -> Kb [M,H] bf16 ; z = 2 -> Vt [B][H][S] bf16 (transposed)
__global__ __launch_bounds__(256) void qkv_gemm(const bf16_t* __restrict__ xb,
                                                const bf16_t* __restrict__ Wb,
                                                const float* __restrict__ bq,
                                                const float* __restrict__ bk,
                                                const float* __restrict__ bv,
                                                bf16_t* __restrict__ Qb,
                                                bf16_t* __restrict__ Kb,
                                                bf16_t* __restrict__ Vt) {
  const int z = blockIdx.z;
  const bf16_t* W = Wb + (size_t)z * HDIM * HDIM;
  const float* bias = (z == 0) ? bq : (z == 1) ? bk : bv;

  const int wave = threadIdx.x >> 5;
  const int lane = threadIdx.x & 31;
  const int m0 = (blockIdx.x * 8 + wave) * 16;
  const int n0 = blockIdx.y * 64;

  v8f acc[4];
#pragma unroll
  for (int j = 0; j < 4; ++j) acc[j] = zero8();

  // 2x-unrolled software pipeline with two disjoint buffer sets (no reg moves)
  v16bf a0 = load_frag_a(xb, HDIM, m0, 0, lane);
  v16bf b0[4];
#pragma unroll
  for (int j = 0; j < 4; ++j) b0[j] = load_frag_bT(W, HDIM, n0 + j * 16, 0, lane);

  for (int kc = 0; kc < HDIM / 32 - 2; kc += 2) {
    const int k1 = (kc + 1) * 32, k2 = (kc + 2) * 32;
    v16bf a1 = load_frag_a(xb, HDIM, m0, k1, lane);
    v16bf b1[4];
#pragma unroll
    for (int j = 0; j < 4; ++j) b1[j] = load_frag_bT(W, HDIM, n0 + j * 16, k1, lane);
#pragma unroll
    for (int j = 0; j < 4; ++j) acc[j] = wmma_bf16(a0, b0[j], acc[j]);
    a0 = load_frag_a(xb, HDIM, m0, k2, lane);
#pragma unroll
    for (int j = 0; j < 4; ++j) b0[j] = load_frag_bT(W, HDIM, n0 + j * 16, k2, lane);
#pragma unroll
    for (int j = 0; j < 4; ++j) acc[j] = wmma_bf16(a1, b1[j], acc[j]);
  }
  {  // tail: chunks 30, 31
    const int k1 = (HDIM / 32 - 1) * 32;
    v16bf a1 = load_frag_a(xb, HDIM, m0, k1, lane);
    v16bf b1[4];
#pragma unroll
    for (int j = 0; j < 4; ++j) b1[j] = load_frag_bT(W, HDIM, n0 + j * 16, k1, lane);
#pragma unroll
    for (int j = 0; j < 4; ++j) acc[j] = wmma_bf16(a0, b0[j], acc[j]);
#pragma unroll
    for (int j = 0; j < 4; ++j) acc[j] = wmma_bf16(a1, b1[j], acc[j]);
  }

  const int half = lane >> 4;
  const int col = lane & 15;
#pragma unroll
  for (int j = 0; j < 4; ++j) {
    const int n = n0 + j * 16 + col;
    const float bi = bias[n];
#pragma unroll
    for (int r = 0; r < 8; ++r) {
      const int m = m0 + r + half * 8;
      bf16_t h = f2bf(acc[j][r] + bi);
      if (z == 0) {
        Qb[(size_t)m * HDIM + n] = h;
      } else if (z == 1) {
        Kb[(size_t)m * HDIM + n] = h;
      } else {
        const int b = m >> 11;       // SEQ = 2048
        const int s = m & (SEQ - 1);
        Vt[((size_t)b * HDIM + n) * SEQ + s] = h;
      }
    }
  }
}

// score masking identical to reference:
//   s = qk/32 ; dead if (k>q) || (s==0) [tril quirk] || !(mask[q]&&mask[k]) ; dead -> -1e9
__device__ __forceinline__ float mask_score(float raw, int k, int q, unsigned qbits,
                                            int row, int mk) {
  float s = raw * 0.03125f;  // 1/sqrt(1024)
  bool ok = (k <= q) && (s != 0.0f) && (((qbits >> row) & 1u) != 0u) && (mk != 0);
  return ok ? s : -1.0e9f;
}

// ---------- kernel 2: softmax row statistics (m, l) over ALL S columns ----------
__global__ __launch_bounds__(256) void attn_stats(const bf16_t* __restrict__ Qb,
                                                  const bf16_t* __restrict__ Kb,
                                                  const int* __restrict__ amask,
                                                  float* __restrict__ Mbuf,
                                                  float* __restrict__ Lbuf) {
  __shared__ bf16_t qtile[16 * HDIM];   // 32 KB, staged via async DMA
  __shared__ float sm[8][16];
  __shared__ float sl[8][16];

  const int b = blockIdx.y;
  const int q0 = blockIdx.x * 16;
  const int wave = threadIdx.x >> 5;
  const int lane = threadIdx.x & 31;
  const int half = lane >> 4;
  const int col = lane & 15;

  const bf16_t* Qrow = Qb + (size_t)b * SEQ * HDIM;
  const bf16_t* Krow = Kb + (size_t)b * SEQ * HDIM;

  // async-stage this block's 16xH Q tile (contiguous 32KB) into LDS
  stage_async(qtile, Qrow + (size_t)q0 * HDIM, 16 * HDIM * 2, threadIdx.x, 256);

  unsigned qbits = 0;
#pragma unroll
  for (int row = 0; row < 16; ++row)
    qbits |= (amask[b * SEQ + q0 + row] != 0 ? 1u : 0u) << row;

  __syncthreads();  // LDS Q tile visible to all waves

  float m[8], l[8];
#pragma unroll
  for (int r = 0; r < 8; ++r) { m[r] = -3.0e38f; l[r] = 0.0f; }

  for (int kt = wave; kt < SEQ / 16; kt += 8) {
    v8f acc = zero8();
    v16bf a0 = load_frag_a(qtile, HDIM, 0, 0, lane);
    v16bf b0 = load_frag_bT(Krow, HDIM, kt * 16, 0, lane);
    for (int kc = 0; kc < HDIM / 32 - 2; kc += 2) {
      const int k1 = (kc + 1) * 32, k2 = (kc + 2) * 32;
      v16bf a1 = load_frag_a(qtile, HDIM, 0, k1, lane);
      v16bf b1 = load_frag_bT(Krow, HDIM, kt * 16, k1, lane);
      acc = wmma_bf16(a0, b0, acc);
      a0 = load_frag_a(qtile, HDIM, 0, k2, lane);
      b0 = load_frag_bT(Krow, HDIM, kt * 16, k2, lane);
      acc = wmma_bf16(a1, b1, acc);
    }
    {
      const int k1 = (HDIM / 32 - 1) * 32;
      v16bf a1 = load_frag_a(qtile, HDIM, 0, k1, lane);
      v16bf b1 = load_frag_bT(Krow, HDIM, kt * 16, k1, lane);
      acc = wmma_bf16(a0, b0, acc);
      acc = wmma_bf16(a1, b1, acc);
    }

    const int kcol = kt * 16 + col;
    const int mk = amask[b * SEQ + kcol];
#pragma unroll
    for (int r = 0; r < 8; ++r) {
      const int row = r + half * 8;
      const int q = q0 + row;
      float s = mask_score(acc[r], kcol, q, qbits, row, mk);
      // row max across the 16 columns (lane groups of 16)
      float rm = s;
#pragma unroll
      for (int off = 1; off < 16; off <<= 1) rm = fmaxf(rm, __shfl_xor(rm, off, 32));
      const float mn = fmaxf(m[r], rm);
      float e = __expf(s - mn);
#pragma unroll
      for (int off = 1; off < 16; off <<= 1) e += __shfl_xor(e, off, 32);
      l[r] = l[r] * __expf(m[r] - mn) + e;
      m[r] = mn;
    }
  }

#pragma unroll
  for (int r = 0; r < 8; ++r) {
    if (col == 0) {
      const int row = r + half * 8;
      sm[wave][row] = m[r];
      sl[wave][row] = l[r];
    }
  }
  __syncthreads();

  if (threadIdx.x < 16) {
    const int row = threadIdx.x;
    float M = -3.0e38f;
#pragma unroll
    for (int w = 0; w < 8; ++w) M = fmaxf(M, sm[w][row]);
    float L = 0.0f;
#pragma unroll
    for (int w = 0; w < 8; ++w) L += sl[w][row] * __expf(sm[w][row] - M);
    Mbuf[b * SEQ + q0 + row] = M;
    Lbuf[b * SEQ + q0 + row] = L;
  }
}

// ---------- kernel 3: output = P @ V  (P recomputed, normalized with m,l) ----------
__global__ __launch_bounds__(256) void attn_out(const bf16_t* __restrict__ Qb,
                                                const bf16_t* __restrict__ Kb,
                                                const bf16_t* __restrict__ Vt,
                                                const int* __restrict__ amask,
                                                const float* __restrict__ Mbuf,
                                                const float* __restrict__ Lbuf,
                                                float* __restrict__ out) {
  __shared__ bf16_t qtile[16 * HDIM];  // 32 KB, staged via async DMA
  __shared__ bf16_t pt[8][16 * 32];    // 8 producer tiles of P (16q x 32k) = 8 KB

  const int b = blockIdx.y;
  const int q0 = blockIdx.x * 16;
  const int wave = threadIdx.x >> 5;
  const int lane = threadIdx.x & 31;
  const int half = lane >> 4;
  const int col = lane & 15;
  const int n0 = wave * 128;

  const bf16_t* Qrow = Qb + (size_t)b * SEQ * HDIM;
  const bf16_t* Krow = Kb + (size_t)b * SEQ * HDIM;
  const bf16_t* Vb   = Vt + (size_t)b * HDIM * SEQ;

  stage_async(qtile, Qrow + (size_t)q0 * HDIM, 16 * HDIM * 2, threadIdx.x, 256);

  unsigned qbits = 0;
#pragma unroll
  for (int row = 0; row < 16; ++row)
    qbits |= (amask[b * SEQ + q0 + row] != 0 ? 1u : 0u) << row;

  float mrow[8], invl[8];
#pragma unroll
  for (int r = 0; r < 8; ++r) {
    const int row = r + half * 8;
    mrow[r] = Mbuf[b * SEQ + q0 + row];
    invl[r] = 1.0f / Lbuf[b * SEQ + q0 + row];
  }

  v8f oacc[8];
#pragma unroll
  for (int j = 0; j < 8; ++j) oacc[j] = zero8();

  __syncthreads();  // LDS Q tile ready

  for (int ro = 0; ro < 8; ++ro) {
    // ---- produce: this wave's 16x32 P tile for keys [kb, kb+32) ----
    const int k32 = ro * 8 + wave;
    const int kb = k32 * 32;
    v8f sA = zero8(), sB = zero8();
    v16bf a0  = load_frag_a(qtile, HDIM, 0, 0, lane);
    v16bf kA0 = load_frag_bT(Krow, HDIM, kb, 0, lane);
    v16bf kB0 = load_frag_bT(Krow, HDIM, kb + 16, 0, lane);
    for (int kc = 0; kc < HDIM / 32 - 2; kc += 2) {
      const int k1 = (kc + 1) * 32, k2 = (kc + 2) * 32;
      v16bf a1  = load_frag_a(qtile, HDIM, 0, k1, lane);
      v16bf kA1 = load_frag_bT(Krow, HDIM, kb, k1, lane);
      v16bf kB1 = load_frag_bT(Krow, HDIM, kb + 16, k1, lane);
      sA = wmma_bf16(a0, kA0, sA);
      sB = wmma_bf16(a0, kB0, sB);
      a0  = load_frag_a(qtile, HDIM, 0, k2, lane);
      kA0 = load_frag_bT(Krow, HDIM, kb, k2, lane);
      kB0 = load_frag_bT(Krow, HDIM, kb + 16, k2, lane);
      sA = wmma_bf16(a1, kA1, sA);
      sB = wmma_bf16(a1, kB1, sB);
    }
    {
      const int k1 = (HDIM / 32 - 1) * 32;
      v16bf a1  = load_frag_a(qtile, HDIM, 0, k1, lane);
      v16bf kA1 = load_frag_bT(Krow, HDIM, kb, k1, lane);
      v16bf kB1 = load_frag_bT(Krow, HDIM, kb + 16, k1, lane);
      sA = wmma_bf16(a0, kA0, sA);
      sB = wmma_bf16(a0, kB0, sB);
      sA = wmma_bf16(a1, kA1, sA);
      sB = wmma_bf16(a1, kB1, sB);
    }

    const int mk0 = amask[b * SEQ + kb + col];
    const int mk1 = amask[b * SEQ + kb + 16 + col];
#pragma unroll
    for (int r = 0; r < 8; ++r) {
      const int row = r + half * 8;
      const int q = q0 + row;
      float s0 = mask_score(sA[r], kb + col,      q, qbits, row, mk0);
      float s1 = mask_score(sB[r], kb + 16 + col, q, qbits, row, mk1);
      float p0 = __expf(s0 - mrow[r]) * invl[r];
      float p1 = __expf(s1 - mrow[r]) * invl[r];
      pt[wave][row * 32 + col]      = f2bf(p0);
      pt[wave][row * 32 + 16 + col] = f2bf(p1);
    }
    __syncthreads();

    // ---- consume: all 8 P tiles against this wave's 128 output columns ----
#pragma unroll
    for (int t = 0; t < 8; ++t) {
      const int kbt = (ro * 8 + t) * 32;
      v16bf a = load_frag_a(&pt[t][0], 32, 0, 0, lane);
      v16bf vb[8];
#pragma unroll
      for (int ns = 0; ns < 8; ++ns)
        vb[ns] = load_frag_bT(Vb, SEQ, n0 + ns * 16, kbt, lane);
#pragma unroll
      for (int ns = 0; ns < 8; ++ns) oacc[ns] = wmma_bf16(a, vb[ns], oacc[ns]);
    }
    __syncthreads();
  }

#pragma unroll
  for (int ns = 0; ns < 8; ++ns) {
#pragma unroll
    for (int r = 0; r < 8; ++r) {
      const int row = r + half * 8;
      out[((size_t)(b * SEQ + q0 + row)) * HDIM + n0 + ns * 16 + col] = oacc[ns][r];
    }
  }
}

// ---------- launcher ----------
extern "C" void kernel_launch(void* const* d_in, const int* in_sizes, int n_in,
                              void* d_out, int out_size, void* d_ws, size_t ws_size,
                              hipStream_t stream) {
  (void)in_sizes; (void)n_in; (void)out_size; (void)ws_size;
  const float* x  = (const float*)d_in[0];
  const int* am   = (const int*)d_in[1];   // attention_mask
  const float* Wq = (const float*)d_in[2];
  const float* bq = (const float*)d_in[3];
  const float* Wk = (const float*)d_in[4];
  const float* bk = (const float*)d_in[5];
  const float* Wv = (const float*)d_in[6];
  const float* bv = (const float*)d_in[7];
  float* out = (float*)d_out;

  // workspace carve-up (~134 MB)
  char* ws = (char*)d_ws;
  const size_t M = (size_t)BATCH * SEQ;            // 16384
  bf16_t* xb = (bf16_t*)ws;             ws += M * HDIM * sizeof(bf16_t);
  bf16_t* Wb = (bf16_t*)ws;             ws += (size_t)3 * HDIM * HDIM * sizeof(bf16_t);
  bf16_t* Qb = (bf16_t*)ws;             ws += M * HDIM * sizeof(bf16_t);
  bf16_t* Kb = (bf16_t*)ws;             ws += M * HDIM * sizeof(bf16_t);
  bf16_t* Vt = (bf16_t*)ws;             ws += M * HDIM * sizeof(bf16_t);
  float* Mbuf = (float*)ws;             ws += M * sizeof(float);
  float* Lbuf = (float*)ws;

  // 0) fp32 -> bf16
  cvt_f32_bf16<<<2048, 256, 0, stream>>>(x, xb, (int)(M * HDIM / 4));
  cvt_f32_bf16<<<512, 256, 0, stream>>>(Wq, Wb + 0 * (size_t)HDIM * HDIM, HDIM * HDIM / 4);
  cvt_f32_bf16<<<512, 256, 0, stream>>>(Wk, Wb + 1 * (size_t)HDIM * HDIM, HDIM * HDIM / 4);
  cvt_f32_bf16<<<512, 256, 0, stream>>>(Wv, Wb + 2 * (size_t)HDIM * HDIM, HDIM * HDIM / 4);

  // 1) Q/K/V projections (V transposed)
  qkv_gemm<<<dim3(M / 16 / 8, HDIM / 64, 3), 256, 0, stream>>>(xb, Wb, bq, bk, bv,
                                                               Qb, Kb, Vt);

  // 2) softmax stats (m, l) per query row
  attn_stats<<<dim3(SEQ / 16, BATCH), 256, 0, stream>>>(Qb, Kb, am, Mbuf, Lbuf);

  // 3) P @ V -> out
  attn_out<<<dim3(SEQ / 16, BATCH), 256, 0, stream>>>(Qb, Kb, Vt, am, Mbuf, Lbuf, out);
}
